// Decoder_44805098832339
// MI455X (gfx1250) — matrix operand order repaired and checked by reading, not compiled, not run on hardware
//
#include <hip/hip_runtime.h>
#include <hip/hip_bf16.h>
#include <cstddef>

// ---------------------------------------------------------------------------
// Problem constants (fixed by the reference)
// ---------------------------------------------------------------------------
#define BB   64      // batch
#define VV   32000   // vocab
#define EE   512     // embedding dim
#define HH   1024    // hidden dim
#define G3   3072    // 3*H
#define TT   64      // steps

typedef __bf16 bf16_t;
typedef __attribute__((ext_vector_type(16))) bf16_t v16bf;
typedef __attribute__((ext_vector_type(8)))  bf16_t v8bf;
typedef __attribute__((ext_vector_type(8)))  float  v8f;

// ---------------------------------------------------------------------------
// helpers
// ---------------------------------------------------------------------------
__device__ __forceinline__ bf16_t f2bf(float f) {
    unsigned u = __builtin_bit_cast(unsigned, f);
    u += 0x7FFFu + ((u >> 16) & 1u);              // round-to-nearest-even
    unsigned short h = (unsigned short)(u >> 16);
    return __builtin_bit_cast(bf16_t, h);
}

__device__ __forceinline__ unsigned wanghash(unsigned x) {
    x = (x ^ 61u) ^ (x >> 16);
    x *= 9u;
    x ^= x >> 4;
    x *= 0x27d4eb2du;
    x ^= x >> 15;
    return x;
}

__device__ __forceinline__ v16bf load_a_frag(const bf16_t* Ap, int kb, int half) {
    v8bf alo = *(const v8bf*)(Ap + kb + half * 16);
    v8bf ahi = *(const v8bf*)(Ap + kb + half * 16 + 8);
    v16bf a;
#pragma unroll
    for (int i = 0; i < 8; ++i) { a[i] = alo[i]; a[i + 8] = ahi[i]; }
    return a;
}

// ---------------------------------------------------------------------------
// f32 -> bf16 weight conversion (runs once per launch; weights then live in
// L2 in bf16: W_ih+W_hh+W_out ~ 75 MB << 192 MB global L2)
// ---------------------------------------------------------------------------
__global__ void k_convert(const float* __restrict__ src,
                          bf16_t* __restrict__ dst, long long n) {
    long long i = (long long)blockIdx.x * blockDim.x + threadIdx.x;
    long long stride = (long long)gridDim.x * blockDim.x;
    for (; i < n; i += stride) dst[i] = f2bf(src[i]);
}

// ---------------------------------------------------------------------------
// init: h0 = concat(init_hidden, att_embedding); x0 = emb[inputs]
// ---------------------------------------------------------------------------
__global__ void k_init(const float* __restrict__ init_hidden,  // (B,768)
                       const float* __restrict__ att_emb,      // (B,256)
                       const int*   __restrict__ tokens,       // (B,1)
                       const float* __restrict__ emb,          // (V,E)
                       float* __restrict__ hF, bf16_t* __restrict__ hB,
                       bf16_t* __restrict__ xB) {
    int idx = blockIdx.x * blockDim.x + threadIdx.x;
    if (idx < BB * HH) {
        int b = idx / HH, d = idx % HH;
        float v = (d < 768) ? init_hidden[b * 768 + d]
                            : att_emb[b * 256 + (d - 768)];
        hF[idx] = v;
        hB[idx] = f2bf(v);
    } else {
        idx -= BB * HH;
        if (idx < BB * EE) {
            int b = idx / EE, e = idx % EE;
            int tok = tokens[b];
            xB[idx] = f2bf(emb[(size_t)tok * EE + e]);
        }
    }
}

// ---------------------------------------------------------------------------
// Gate GEMMs: gi = x @ W_ih^T + b_ih   (K=512)   [blockIdx.y == 0]
//             gh = h @ W_hh^T + b_hh   (K=1024)  [blockIdx.y == 1]
// 128 threads = 4 waves; wave w owns M-tile w (M=64); block owns 32 cols
// (2-way N register blocking: one A fragment feeds 2 WMMAs per K-step).
// ---------------------------------------------------------------------------
__global__ void k_gates(const bf16_t* __restrict__ xB,
                        const bf16_t* __restrict__ hB,
                        const bf16_t* __restrict__ wihB,
                        const bf16_t* __restrict__ whhB,
                        const float*  __restrict__ b_ih,
                        const float*  __restrict__ b_hh,
                        float* __restrict__ gi, float* __restrict__ gh) {
    const int sel = blockIdx.y;                    // uniform per block
    const bf16_t* A;  const bf16_t* W;  const float* bias;  float* out;  int K;
    if (sel == 0) { A = xB; W = wihB; bias = b_ih; out = gi; K = EE; }
    else          { A = hB; W = whhB; bias = b_hh; out = gh; K = HH; }

    const int lane = threadIdx.x & 31;
    const int wv   = threadIdx.x >> 5;             // M-tile id (0..3)
    const int m16  = lane & 15;
    const int half = lane >> 4;                    // K-half selector
    const int n0   = blockIdx.x * 32;

    const bf16_t* Ap  = A + (size_t)(wv * 16 + m16) * K;
    const bf16_t* Bp0 = W + (size_t)(n0 + m16) * K + half * 16;
    const bf16_t* Bp1 = Bp0 + (size_t)16 * K;

    v8f acc0 = {}, acc1 = {};
    for (int kb = 0; kb < K; kb += 32) {
        v16bf a  = load_a_frag(Ap, kb, half);
        v16bf b0 = *(const v16bf*)(Bp0 + kb);
        v16bf b1 = *(const v16bf*)(Bp1 + kb);
        acc0 = __builtin_amdgcn_wmma_f32_16x16x32_bf16(
            false, a, false, b0, (short)0, acc0, false, false);
        acc1 = __builtin_amdgcn_wmma_f32_16x16x32_bf16(
            false, a, false, b1, (short)0, acc1, false, false);
    }

    const int col0 = n0 + m16;
    const float bv0 = bias[col0];
    const float bv1 = bias[col0 + 16];
#pragma unroll
    for (int v = 0; v < 8; ++v) {
        int row = wv * 16 + v + half * 8;
        out[(size_t)row * G3 + col0]      = acc0[v] + bv0;
        out[(size_t)row * G3 + col0 + 16] = acc1[v] + bv1;
    }
}

// ---------------------------------------------------------------------------
// GRU elementwise update
// ---------------------------------------------------------------------------
__global__ void k_gru(const float* __restrict__ gi,
                      const float* __restrict__ gh,
                      float* __restrict__ hF, bf16_t* __restrict__ hB) {
    int idx = blockIdx.x * blockDim.x + threadIdx.x;   // over B*H
    if (idx >= BB * HH) return;
    int b = idx / HH, j = idx % HH;
    size_t base = (size_t)b * G3;
    float ir = gi[base + j], iz = gi[base + HH + j], in = gi[base + 2 * HH + j];
    float hr = gh[base + j], hz = gh[base + HH + j], hn = gh[base + 2 * HH + j];
    float r = 1.0f / (1.0f + __expf(-(ir + hr)));
    float z = 1.0f / (1.0f + __expf(-(iz + hz)));
    float n = tanhf(in + r * hn);
    float h = hF[idx];
    float hnw = (1.0f - z) * n + z * h;
    hF[idx] = hnw;
    hB[idx] = f2bf(hnw);
}

// ---------------------------------------------------------------------------
// Output GEMM: logits[t] = h @ W_out^T + b_out  (64 x 32000, K=1024)
// 500 blocks x 128 threads; each wave holds one A fragment and drives
// 4 WMMAs against 4 B tiles (64 columns per block) -> load:wmma = 2.5:1,
// 4 independent accumulators per K-step hide WMMA latency.
// ---------------------------------------------------------------------------
__global__ void k_logits(const bf16_t* __restrict__ hB,
                         const bf16_t* __restrict__ woutB,
                         const float*  __restrict__ b_out,
                         float* __restrict__ logitsT) {  // base for this step
    const int lane = threadIdx.x & 31;
    const int wv   = threadIdx.x >> 5;
    const int m16  = lane & 15;
    const int half = lane >> 4;
    const int n0   = blockIdx.x * 64;

    const bf16_t* Ap  = hB + (size_t)(wv * 16 + m16) * HH;
    const bf16_t* Bp0 = woutB + (size_t)(n0 + m16) * HH + half * 16;
    const bf16_t* Bp1 = Bp0 + (size_t)16 * HH;
    const bf16_t* Bp2 = Bp0 + (size_t)32 * HH;
    const bf16_t* Bp3 = Bp0 + (size_t)48 * HH;

    v8f acc0 = {}, acc1 = {}, acc2 = {}, acc3 = {};
    for (int kb = 0; kb < HH; kb += 32) {
        v16bf a  = load_a_frag(Ap, kb, half);
        v16bf b0 = *(const v16bf*)(Bp0 + kb);
        v16bf b1 = *(const v16bf*)(Bp1 + kb);
        v16bf b2 = *(const v16bf*)(Bp2 + kb);
        v16bf b3 = *(const v16bf*)(Bp3 + kb);
        acc0 = __builtin_amdgcn_wmma_f32_16x16x32_bf16(
            false, a, false, b0, (short)0, acc0, false, false);
        acc1 = __builtin_amdgcn_wmma_f32_16x16x32_bf16(
            false, a, false, b1, (short)0, acc1, false, false);
        acc2 = __builtin_amdgcn_wmma_f32_16x16x32_bf16(
            false, a, false, b2, (short)0, acc2, false, false);
        acc3 = __builtin_amdgcn_wmma_f32_16x16x32_bf16(
            false, a, false, b3, (short)0, acc3, false, false);
    }

    const int col = n0 + m16;
    const float bv0 = b_out[col];
    const float bv1 = b_out[col + 16];
    const float bv2 = b_out[col + 32];
    const float bv3 = b_out[col + 48];
#pragma unroll
    for (int v = 0; v < 8; ++v) {
        int row = wv * 16 + v + half * 8;
        float* r = logitsT + (size_t)row * VV + col;
        r[0]  = acc0[v] + bv0;
        r[16] = acc1[v] + bv1;
        r[32] = acc2[v] + bv2;
        r[48] = acc3[v] + bv3;
    }
}

// ---------------------------------------------------------------------------
// Gumbel-argmax sampling (deterministic hash RNG) + embedding gather for the
// next step's x. One block per batch row.
// ---------------------------------------------------------------------------
__global__ void k_sample(const float* __restrict__ logitsT,  // (64, 32000)
                         const float* __restrict__ emb,      // (V, E) f32
                         bf16_t* __restrict__ xB,            // (B, E) bf16
                         float*  __restrict__ idsOut,        // (B, T) region
                         int t) {
    const int b = blockIdx.x;
    const int tid = threadIdx.x;
    const float* L = logitsT + (size_t)b * VV;

    float best = -3.4e38f;
    int   bi   = 0;
    const unsigned salt = 0x9E3779B9u * (unsigned)(t * BB + b);
    for (int v = tid; v < VV; v += 256) {
        unsigned h = wanghash((unsigned)v ^ salt);
        float u = ((h >> 8) + 0.5f) * 5.9604645e-08f;      // (0,1)
        float g = -__logf(-__logf(u));                     // Gumbel(0,1)
        float tot = L[v] + g;
        if (tot > best || (tot == best && v < bi)) { best = tot; bi = v; }
    }

    __shared__ float sv[256];
    __shared__ int   si[256];
    sv[tid] = best; si[tid] = bi;
    __syncthreads();
    for (int s = 128; s > 0; s >>= 1) {
        if (tid < s) {
            if (sv[tid + s] > sv[tid] ||
                (sv[tid + s] == sv[tid] && si[tid + s] < si[tid])) {
                sv[tid] = sv[tid + s];
                si[tid] = si[tid + s];
            }
        }
        __syncthreads();
    }
    const int sid = si[0];
    if (tid == 0) idsOut[b * TT + t] = (float)sid;

    for (int e = tid; e < EE; e += 256)
        xB[(size_t)b * EE + e] = f2bf(emb[(size_t)sid * EE + e]);
}

// ---------------------------------------------------------------------------
// host-side orchestration
// ---------------------------------------------------------------------------
extern "C" void kernel_launch(void* const* d_in, const int* in_sizes, int n_in,
                              void* d_out, int out_size, void* d_ws, size_t ws_size,
                              hipStream_t stream) {
    const int*   inputs      = (const int*)  d_in[0];
    // d_in[1] = max_length (compile-time TT)
    const float* init_hidden = (const float*)d_in[2];
    const float* att_emb     = (const float*)d_in[3];
    const float* emb         = (const float*)d_in[4];
    const float* W_ih        = (const float*)d_in[5];
    const float* W_hh        = (const float*)d_in[6];
    const float* b_ih        = (const float*)d_in[7];
    const float* b_hh        = (const float*)d_in[8];
    const float* W_out       = (const float*)d_in[9];
    const float* b_out       = (const float*)d_in[10];

    float* out    = (float*)d_out;
    float* logits = out;                                  // (T*B, V)
    float* idsOut = out + (size_t)TT * BB * VV;           // (B, T)

    // workspace layout (all chunks 256B-aligned by construction)
    char* w = (char*)d_ws;
    bf16_t* wihB  = (bf16_t*)w;  w += (size_t)G3 * EE * sizeof(bf16_t);
    bf16_t* whhB  = (bf16_t*)w;  w += (size_t)G3 * HH * sizeof(bf16_t);
    bf16_t* woutB = (bf16_t*)w;  w += (size_t)VV * HH * sizeof(bf16_t);
    float*  gi    = (float*) w;  w += (size_t)BB * G3 * sizeof(float);
    float*  gh    = (float*) w;  w += (size_t)BB * G3 * sizeof(float);
    float*  hF    = (float*) w;  w += (size_t)BB * HH * sizeof(float);
    bf16_t* hB    = (bf16_t*)w;  w += (size_t)BB * HH * sizeof(bf16_t);
    bf16_t* xB    = (bf16_t*)w;  w += (size_t)BB * EE * sizeof(bf16_t);

    // one-time bf16 weight staging (then L2-resident across all 64 steps)
    k_convert<<<2048, 256, 0, stream>>>(W_ih,  wihB,  (long long)G3 * EE);
    k_convert<<<2048, 256, 0, stream>>>(W_hh,  whhB,  (long long)G3 * HH);
    k_convert<<<8192, 256, 0, stream>>>(W_out, woutB, (long long)VV * HH);

    // h0 / x0
    k_init<<<(BB * HH + BB * EE + 255) / 256, 256, 0, stream>>>(
        init_hidden, att_emb, inputs, emb, hF, hB, xB);

    // sequential recurrence, ordered by the stream
    for (int t = 0; t < TT; ++t) {
        k_gates<<<dim3(G3 / 32, 2), 128, 0, stream>>>(
            xB, hB, wihB, whhB, b_ih, b_hh, gi, gh);
        k_gru<<<(BB * HH + 255) / 256, 256, 0, stream>>>(gi, gh, hF, hB);
        float* logitsT = logits + (size_t)t * BB * VV;
        k_logits<<<VV / 64, 128, 0, stream>>>(hB, woutB, b_out, logitsT);
        k_sample<<<BB, 256, 0, stream>>>(logitsT, emb, xB, idsOut, t);
    }
}
